// SymODEN_R1_T1_27221502722062
// MI455X (gfx1250) — matrix operand order repaired and checked by reading, not compiled, not run on hardware
//
#include <hip/hip_runtime.h>

typedef __attribute__((ext_vector_type(16))) __bf16 v16bf;
typedef __attribute__((ext_vector_type(8)))  __bf16 v8bf;
typedef __attribute__((ext_vector_type(8)))  float  v8f;

constexpr int HIDN = 128;

// ---------------- LDS layout (dynamic shared) ----------------
struct alignas(16) LdsLayout {
  __bf16 mW2T[HIDN * HIDN];           // [n][k]  (transposed, forward B-operand)
  __bf16 gW2T[HIDN * HIDN];           // [n][k]
  __bf16 hW2T[HIDN * HIDN];           // [n][k]
  __bf16 hW2N[HIDN * HIDN];           // [k][n] row-major original -> B-operand for W2^T (backward)
  __bf16 scratch[8][3][16 * HIDN];    // per-wave transpose staging / persistent activations
  float mW1[3 * HIDN]; float mB1[HIDN]; float mB2[HIDN]; float mW3[HIDN * 4]; float mB3[4];
  float hW1[5 * HIDN]; float hW1T[HIDN * 5]; float hB1[HIDN]; float hB2[HIDN]; float hW3[HIDN];
  float gW1[3 * HIDN]; float gB1[HIDN]; float gB2[HIDN]; float gW3[HIDN * 2]; float gB3[2];
};

// ---------------- helpers ----------------
__device__ __forceinline__ float fast_tanh(float x) {
#if __has_builtin(__builtin_amdgcn_tanhf)
  return __builtin_amdgcn_tanhf(x);          // native v_tanh_f32 (TRANS32, co-executes with WMMA)
#elif __has_builtin(__builtin_amdgcn_tanh_f32)
  return __builtin_amdgcn_tanh_f32(x);
#else
  x = fminf(10.0f, fmaxf(-10.0f, x));
  float e = __expf(2.0f * x);
  return (e - 1.0f) * __builtin_amdgcn_rcpf(e + 1.0f);
#endif
}

// Layer-1: tiny in_dim -> 128, computed straight into bf16 A-operand layout.
// A-operand element j of chunk kt (lane l, hi=l>>4): feature f = kt*32 + h2*16 + hi*8 + e
template <int IN>
__device__ __forceinline__ void layer1(const float* x, const float* W1, const float* b1,
                                       int hi, v16bf a[4]) {
#pragma unroll
  for (int kt = 0; kt < 4; ++kt) {
#pragma unroll
    for (int h2 = 0; h2 < 2; ++h2) {
      const int bf = kt * 32 + h2 * 16 + hi * 8;
      float z[8];
#pragma unroll
      for (int e = 0; e < 8; ++e) z[e] = b1[bf + e];
#pragma unroll
      for (int j = 0; j < IN; ++j) {
        const float xj = x[j];
#pragma unroll
        for (int e = 0; e < 8; ++e) z[e] = fmaf(xj, W1[j * HIDN + bf + e], z[e]);
      }
#pragma unroll
      for (int e = 0; e < 8; ++e) a[kt][h2 * 8 + e] = (__bf16)fast_tanh(z[e]);
    }
  }
}

// Tangent layer-1: t1 = (xd @ W1) * (1 - a1^2), no bias/tanh, into A-operand layout.
__device__ __forceinline__ void layer1_tangent3(const float* xd, const float* W1, int hi,
                                                const v16bf a1[4], v16bf t[4]) {
#pragma unroll
  for (int kt = 0; kt < 4; ++kt) {
#pragma unroll
    for (int h2 = 0; h2 < 2; ++h2) {
      const int bf = kt * 32 + h2 * 16 + hi * 8;
      float z[8];
#pragma unroll
      for (int e = 0; e < 8; ++e) z[e] = 0.0f;
#pragma unroll
      for (int j = 0; j < 3; ++j) {
        const float xj = xd[j];
#pragma unroll
        for (int e = 0; e < 8; ++e) z[e] = fmaf(xj, W1[j * HIDN + bf + e], z[e]);
      }
#pragma unroll
      for (int e = 0; e < 8; ++e) {
        float ap = (float)a1[kt][h2 * 8 + e];
        t[kt][h2 * 8 + e] = (__bf16)(z[e] * (1.0f - ap * ap));
      }
    }
  }
}

// 16x128 @ 128x128 via 8 N-tiles x 4 K-chunks of V_WMMA_F32_16X16X32_BF16.
// WT is bf16 [n][k] (k contiguous). All 4 B operands of an N-tile are preloaded
// before the WMMA chain so DS latency pipelines (dscnt<=6/4/2/0 instead of 0x0
// before every WMMA). C starts as inline zero; bias folds into the next elementwise.
__device__ __forceinline__ void matmul128(const v16bf a[4], const __bf16* __restrict__ WT,
                                          int l, v8f acc[8]) {
  const int n15 = l & 15, hi = l >> 4;
#pragma unroll
  for (int nt = 0; nt < 8; ++nt) {
    const __bf16* base = WT + (nt * 16 + n15) * HIDN + hi * 8;
    const v16bf b0 = __builtin_shufflevector(*(const v8bf*)(base), *(const v8bf*)(base + 16), 0, 1, 2, 3, 4, 5, 6, 7, 8, 9, 10, 11, 12, 13, 14, 15);
    const v16bf b1 = __builtin_shufflevector(*(const v8bf*)(base + 32), *(const v8bf*)(base + 48), 0, 1, 2, 3, 4, 5, 6, 7, 8, 9, 10, 11, 12, 13, 14, 15);
    const v16bf b2 = __builtin_shufflevector(*(const v8bf*)(base + 64), *(const v8bf*)(base + 80), 0, 1, 2, 3, 4, 5, 6, 7, 8, 9, 10, 11, 12, 13, 14, 15);
    const v16bf b3 = __builtin_shufflevector(*(const v8bf*)(base + 96), *(const v8bf*)(base + 112), 0, 1, 2, 3, 4, 5, 6, 7, 8, 9, 10, 11, 12, 13, 14, 15);
    v8f c = {};  // zeroinitializer -> SRC2 = inline 0 on first WMMA
    c = __builtin_amdgcn_wmma_f32_16x16x32_bf16(false, a[0], false, b0, (short)0, c, false, false);
    c = __builtin_amdgcn_wmma_f32_16x16x32_bf16(false, a[1], false, b1, (short)0, c, false, false);
    c = __builtin_amdgcn_wmma_f32_16x16x32_bf16(false, a[2], false, b2, (short)0, c, false, false);
    c = __builtin_amdgcn_wmma_f32_16x16x32_bf16(false, a[3], false, b3, (short)0, c, false, false);
    acc[nt] = c;
  }
}

// Row-major bf16 [16][128] scratch -> A-operand registers (two 16B DS loads per chunk).
__device__ __forceinline__ void load_aops(const __bf16* S, int m, int hi, v16bf a[4]) {
#pragma unroll
  for (int kt = 0; kt < 4; ++kt) {
    v8bf lo = *(const v8bf*)(S + m * HIDN + kt * 32 + hi * 8);
    v8bf hh = *(const v8bf*)(S + m * HIDN + kt * 32 + 16 + hi * 8);
    a[kt] = __builtin_shufflevector(lo, hh, 0, 1, 2, 3, 4, 5, 6, 7, 8, 9, 10, 11, 12, 13, 14, 15);
  }
}

// A-operand registers -> row-major bf16 scratch (complementary feature sets per half-wave).
__device__ __forceinline__ void store_aops(__bf16* S, int m, int hi, const v16bf a[4]) {
#pragma unroll
  for (int kt = 0; kt < 4; ++kt) {
    v8bf lo = __builtin_shufflevector(a[kt], a[kt], 0, 1, 2, 3, 4, 5, 6, 7);
    v8bf hh = __builtin_shufflevector(a[kt], a[kt], 8, 9, 10, 11, 12, 13, 14, 15);
    *(v8bf*)(S + m * HIDN + kt * 32 + hi * 8) = lo;
    *(v8bf*)(S + m * HIDN + kt * 32 + 16 + hi * 8) = hh;
  }
}

// C-layout accumulators -> row-major bf16 scratch: tanh(acc + bias[col]).
__device__ __forceinline__ void store_acc_tanh_bias(__bf16* S, const v8f acc[8], int l,
                                                    const float* __restrict__ bias) {
  const int n15 = l & 15, hi = l >> 4;
#pragma unroll
  for (int nt = 0; nt < 8; ++nt) {
    const float bv = bias[nt * 16 + n15];
#pragma unroll
    for (int r = 0; r < 8; ++r)
      S[(r + hi * 8) * HIDN + nt * 16 + n15] = (__bf16)fast_tanh(acc[nt][r] + bv);
  }
}
__device__ __forceinline__ void store_acc_raw(__bf16* S, const v8f acc[8], int l) {
  const int n15 = l & 15, hi = l >> 4;
#pragma unroll
  for (int nt = 0; nt < 8; ++nt)
#pragma unroll
    for (int r = 0; r < 8; ++r)
      S[(r + hi * 8) * HIDN + nt * 16 + n15] = (__bf16)acc[nt][r];
}

// Per-sample small dot, split across half-waves (hi half sums features hi*64..hi*64+63)
// and combined with one cross-half shuffle per output: out[d] = sum_f S[m][f] * W[f*DOUT + d]
template <int DOUT>
__device__ __forceinline__ void rowdot(const __bf16* S, int m, int hi,
                                       const float* __restrict__ W, float out[DOUT]) {
#pragma unroll
  for (int d = 0; d < DOUT; ++d) out[d] = 0.0f;
#pragma unroll
  for (int c = 0; c < 8; ++c) {
    const int cc = hi * 8 + c;
    v8bf v = *(const v8bf*)(S + m * HIDN + cc * 8);
#pragma unroll
    for (int e = 0; e < 8; ++e) {
      float a = (float)v[e];
#pragma unroll
      for (int d = 0; d < DOUT; ++d) out[d] = fmaf(a, W[(cc * 8 + e) * DOUT + d], out[d]);
    }
  }
#pragma unroll
  for (int d = 0; d < DOUT; ++d) out[d] += __shfl_xor(out[d], 16, 32);
}

// out[d] = sum_f raw[m][f] * (1 - mask[m][f]^2) * W[f*DOUT + d], half-wave split as above.
// (mask holds post-tanh activations; applies the tanh' chain-rule factor during the dot)
template <int DOUT>
__device__ __forceinline__ void rowdot_masked(const __bf16* Sraw, const __bf16* Smask, int m,
                                              int hi, const float* __restrict__ W,
                                              float out[DOUT]) {
#pragma unroll
  for (int d = 0; d < DOUT; ++d) out[d] = 0.0f;
#pragma unroll
  for (int c = 0; c < 8; ++c) {
    const int cc = hi * 8 + c;
    v8bf vr = *(const v8bf*)(Sraw + m * HIDN + cc * 8);
    v8bf vm = *(const v8bf*)(Smask + m * HIDN + cc * 8);
#pragma unroll
    for (int e = 0; e < 8; ++e) {
      float s = (float)vm[e];
      float a = (float)vr[e] * (1.0f - s * s);
#pragma unroll
      for (int d = 0; d < DOUT; ++d) out[d] = fmaf(a, W[(cc * 8 + e) * DOUT + d], out[d]);
    }
  }
#pragma unroll
  for (int d = 0; d < DOUT; ++d) out[d] += __shfl_xor(out[d], 16, 32);
}

// ---------------- fused SymODEN kernel ----------------
__global__ __launch_bounds__(256) void symoden_kernel(
    const float* __restrict__ y,
    const float* hW1g, const float* hB1g, const float* hW2g, const float* hB2g,
    const float* hW3g, const float* hB3g,
    const float* mW1g, const float* mB1g, const float* mW2g, const float* mB2g,
    const float* mW3g, const float* mB3g,
    const float* gW1g, const float* gB1g, const float* gW2g, const float* gB2g,
    const float* gW3g, const float* gB3g,
    float* __restrict__ out) {
  extern __shared__ char lds_raw[];
  LdsLayout& L = *reinterpret_cast<LdsLayout*>(lds_raw);
  const int tid = threadIdx.x;

  // ---- stage weights into LDS (bf16 for WMMA operands, f32 for small layers) ----
  for (int i = tid; i < HIDN * HIDN; i += 256) {
    const int k = i >> 7, n = i & 127;
    L.mW2T[n * HIDN + k] = (__bf16)mW2g[i];
    L.gW2T[n * HIDN + k] = (__bf16)gW2g[i];
    const float wh = hW2g[i];
    L.hW2T[n * HIDN + k] = (__bf16)wh;
    L.hW2N[i] = (__bf16)wh;
  }
  for (int i = tid; i < 3 * HIDN; i += 256) { L.mW1[i] = mW1g[i]; L.gW1[i] = gW1g[i]; }
  for (int i = tid; i < 5 * HIDN; i += 256) L.hW1[i] = hW1g[i];
  for (int i = tid; i < HIDN * 4; i += 256) L.mW3[i] = mW3g[i];
  for (int i = tid; i < HIDN * 2; i += 256) L.gW3[i] = gW3g[i];
  for (int i = tid; i < HIDN; i += 256) {
    L.mB1[i] = mB1g[i]; L.mB2[i] = mB2g[i];
    L.hB1[i] = hB1g[i]; L.hB2[i] = hB2g[i];
    L.gB1[i] = gB1g[i]; L.gB2[i] = gB2g[i];
    L.hW3[i] = hW3g[i];
#pragma unroll
    for (int j = 0; j < 5; ++j) L.hW1T[i * 5 + j] = hW1g[j * HIDN + i];
  }
  if (tid < 4) L.mB3[tid] = mB3g[tid];
  if (tid < 2) L.gB3[tid] = gB3g[tid];
  __syncthreads();

  const int wave = tid >> 5;
  const int l = tid & 31;
  const int hi = l >> 4;
  const int m = l & 15;
  const int n15 = m;
  __bf16* R0 = L.scratch[wave][0];
  __bf16* R1 = L.scratch[wave][1];
  __bf16* R2 = L.scratch[wave][2];

  const int s = blockIdx.x * 128 + wave * 16 + m;  // sample (duplicated across half-waves)
  const float y0 = y[s * 6 + 0], cq = y[s * 6 + 1], sq = y[s * 6 + 2];
  const float xd0 = y[s * 6 + 3], xd1 = y[s * 6 + 4], uu = y[s * 6 + 5];
  const float q3[3] = {y0, cq, sq};

  v8f acc[8];

  // ======== m-net forward: keep a1m (A-ops, regs) and a2m (row-major, R1) ========
  v16bf a1m[4];
  layer1<3>(q3, L.mW1, L.mB1, hi, a1m);
  matmul128(a1m, L.mW2T, l, acc);
  store_acc_tanh_bias(R1, acc, l, L.mB2);           // a2m -> R1 (persists for tangent mask)
  float A4[4];
  rowdot<4>(R1, m, hi, L.mW3, A4);
#pragma unroll
  for (int d = 0; d < 4; ++d) A4[d] += L.mB3[d];

  // M_q_inv = A A^T + I (symmetric 2x2); p = inv(M_q_inv) @ xq_dot
  const float m00 = A4[0] * A4[0] + A4[1] * A4[1] + 1.0f;
  const float m01 = A4[0] * A4[2] + A4[1] * A4[3];
  const float m11 = A4[2] * A4[2] + A4[3] * A4[3] + 1.0f;
  const float rdet = __builtin_amdgcn_rcpf(m00 * m11 - m01 * m01);
  const float p0 = (m11 * xd0 - m01 * xd1) * rdet;
  const float p1 = (m00 * xd1 - m01 * xd0) * rdet;

  // ======== h-net forward + input-gradient ========
  const float zp[5] = {y0, cq, sq, p0, p1};
  v16bf a1h[4];
  layer1<5>(zp, L.hW1, L.hB1, hi, a1h);
  store_aops(R0, m, hi, a1h);                       // a1h row-major -> R0 (backward mask)
  matmul128(a1h, L.hW2T, l, acc);
  // g2 = W3h[o] * (1 - tanh(acc + b2)^2) -> R2 (bf16, row-major)
#pragma unroll
  for (int nt = 0; nt < 8; ++nt) {
    const float w3v = L.hW3[nt * 16 + n15];
    const float bv = L.hB2[nt * 16 + n15];
#pragma unroll
    for (int r = 0; r < 8; ++r) {
      const float t = fast_tanh(acc[nt][r] + bv);
      R2[(r + hi * 8) * HIDN + nt * 16 + n15] = (__bf16)(w3v * (1.0f - t * t));
    }
  }
  v16bf g2op[4];
  load_aops(R2, m, hi, g2op);
  matmul128(g2op, L.hW2N, l, acc);                  // g2 @ W2h^T  (raw, mask applied in dot)
  store_acc_raw(R2, acc, l);
  float dH[5];
  rowdot_masked<5>(R2, R0, m, hi, L.hW1T, dH);      // dH = (g1post .* (1-a1h^2)) @ W1h^T

  // ======== g-net forward ========
  v16bf a1g[4];
  layer1<3>(q3, L.gW1, L.gB1, hi, a1g);
  matmul128(a1g, L.gW2T, l, acc);
  store_acc_tanh_bias(R0, acc, l, L.gB2);           // a2g -> R0 (a1h dead now)
  float og[2];
  rowdot<2>(R0, m, hi, L.gW3, og);
  const float F0 = (og[0] + L.gB3[0]) * uu;
  const float F1 = (og[1] + L.gB3[1]) * uu;

  // ======== Poisson assembly ========
  const float dHdq = -sq * dH[1] + cq * dH[2];
  const float dx = dH[3], dq = dH[4];
  const float dp0 = -dH[0] + F0;
  const float dp1 = -dHdq + F1;
  const float dq3dt[3] = {dx, -sq * dq, cq * dq};

  // ======== m-net tangent (JVP along dq3dt) ========
  v16bf t1[4];
  layer1_tangent3(dq3dt, L.mW1, hi, a1m, t1);       // mask from cached a1m regs
  matmul128(t1, L.mW2T, l, acc);
  store_acc_raw(R0, acc, l);                        // raw pre-mask tangent -> R0
  float tA[4];
  rowdot_masked<4>(R0, R1, m, hi, L.mW3, tA);       // apply (1 - a2m^2) during W3 dot

  // dM_inv_dt = tA A^T + A tA^T
  const float dm00 = 2.0f * (A4[0] * tA[0] + A4[1] * tA[1]);
  const float dm01 = A4[0] * tA[2] + A4[1] * tA[3] + tA[0] * A4[2] + tA[1] * A4[3];
  const float dm11 = 2.0f * (A4[2] * tA[2] + A4[3] * tA[3]);

  const float ddq0 = m00 * dp0 + m01 * dp1 + dm00 * p0 + dm01 * p1;
  const float ddq1 = m01 * dp0 + m11 * dp1 + dm01 * p0 + dm11 * p1;

  if (hi == 0) {
    float* o = out + (size_t)s * 6;
    o[0] = dx;
    o[1] = -sq * dq;
    o[2] = cq * dq;
    o[3] = ddq0;
    o[4] = ddq1;
    o[5] = 0.0f;
  }
}

// ---------------- launcher ----------------
extern "C" void kernel_launch(void* const* d_in, const int* in_sizes, int n_in,
                              void* d_out, int out_size, void* d_ws, size_t ws_size,
                              hipStream_t stream) {
  (void)in_sizes; (void)n_in; (void)d_ws; (void)ws_size; (void)out_size;
  const float* y = (const float*)d_in[1];
  const float* hW1 = (const float*)d_in[2];
  const float* hB1 = (const float*)d_in[3];
  const float* hW2 = (const float*)d_in[4];
  const float* hB2 = (const float*)d_in[5];
  const float* hW3 = (const float*)d_in[6];
  const float* hB3 = (const float*)d_in[7];
  const float* mW1 = (const float*)d_in[8];
  const float* mB1 = (const float*)d_in[9];
  const float* mW2 = (const float*)d_in[10];
  const float* mB2 = (const float*)d_in[11];
  const float* mW3 = (const float*)d_in[12];
  const float* mB3 = (const float*)d_in[13];
  const float* gW1 = (const float*)d_in[14];
  const float* gB1 = (const float*)d_in[15];
  const float* gW2 = (const float*)d_in[16];
  const float* gB2 = (const float*)d_in[17];
  const float* gW3 = (const float*)d_in[18];
  const float* gB3 = (const float*)d_in[19];

  const size_t shmem = sizeof(LdsLayout);
  (void)hipFuncSetAttribute((const void*)symoden_kernel,
                            hipFuncAttributeMaxDynamicSharedMemorySize, (int)shmem);

  const int BS = 131072;
  const int blocks = BS / 128;  // 8 waves x 16 samples per block
  symoden_kernel<<<blocks, 256, shmem, stream>>>(
      y, hW1, hB1, hW2, hB2, hW3, hB3,
      mW1, mB1, mW2, mB2, mW3, mB3,
      gW1, gB1, gW2, gB2, gW3, gB3,
      (float*)d_out);
}
